// InterLevelAttention_65584150610225
// MI455X (gfx1250) — matrix operand order repaired and checked by reading, not compiled, not run on hardware
//
#include <hip/hip_runtime.h>
#include <hip/hip_bf16.h>

// CDNA5 / gfx1250: wave32, WMMA 16x16x32 bf16 (f32 accumulate).
typedef __attribute__((ext_vector_type(16))) __bf16 v16bf_t;
typedef __attribute__((ext_vector_type(8)))  float  v8f_t;

#define ATTN_DIM   256
#define ATTN_KVDIM 512
#define ATTN_HD    32
#define ATTN_SCALE 0.17677669529663687f  /* 32^-0.5 */

__device__ __forceinline__ unsigned short f32_to_bf16_rne(float f) {
  unsigned u = __float_as_uint(f);
  u += 0x7FFFu + ((u >> 16) & 1u);   // round-to-nearest-even
  return (unsigned short)(u >> 16);
}

union WmmaFrag {
  unsigned short us[16];
  uint4 q[2];
  v16bf_t v;
};

// ---------------------------------------------------------------------------
// f32 -> bf16 conversion (grid-stride)
// ---------------------------------------------------------------------------
__global__ void cvt_f32_to_bf16(const float* __restrict__ in,
                                unsigned short* __restrict__ out, int n) {
  int i = blockIdx.x * blockDim.x + threadIdx.x;
  int stride = gridDim.x * blockDim.x;
  for (; i < n; i += stride) out[i] = f32_to_bf16_rne(in[i]);
}

// ---------------------------------------------------------------------------
// C[M,N] = A[M,K](bf16, row-major) x W[N,K](bf16, row-major, i.e. B=W^T) + bias
// One wave -> 16x64 output stripe (A fragment reused across 4 WMMAs).
// Fragment layout used here (wave32; contiguous K, not the ISA table order):
//   A: lane L holds row L%16; K in {0..15} (L<16) / {16..31}
//   B: lane L holds col L%16; K in {0..15} (L<16) / {16..31}
//   C/D: VGPR r -> row r (L<16) / r+8 (L>=16), col L%16
// ---------------------------------------------------------------------------
__global__ void __launch_bounds__(256)
gemm_bf16_wmma(const unsigned short* __restrict__ A,
               const unsigned short* __restrict__ W,
               const float* __restrict__ bias,
               float* __restrict__ C,
               int M, int N, int K) {
  const int wave = (int)((blockIdx.x * blockDim.x + threadIdx.x) >> 5);
  const int lane = (int)(threadIdx.x & 31u);
  const int nstripes = N >> 6;                 // 64-wide column stripes
  const int tm = (wave / nstripes) << 4;
  const int tn = (wave % nstripes) << 6;
  if (tm >= M) return;
  const int hv = lane >> 4;                    // lane-half (0/1)
  const int lm = lane & 15;

  const unsigned short* arow = A + (size_t)(tm + lm) * K + hv * 16;
  const unsigned short* w0 = W + (size_t)(tn +  0 + lm) * K + hv * 16;
  const unsigned short* w1 = W + (size_t)(tn + 16 + lm) * K + hv * 16;
  const unsigned short* w2 = W + (size_t)(tn + 32 + lm) * K + hv * 16;
  const unsigned short* w3 = W + (size_t)(tn + 48 + lm) * K + hv * 16;

  v8f_t acc0 = {}, acc1 = {}, acc2 = {}, acc3 = {};

  for (int k0 = 0; k0 < K; k0 += 32) {
    WmmaFrag a, b0, b1, b2, b3;
    a.q[0]  = *(const uint4*)(arow + k0);        // K: k0 + hv*16 + 0..7
    a.q[1]  = *(const uint4*)(arow + k0 + 8);    // K: k0 + hv*16 + 8..15
    b0.q[0] = *(const uint4*)(w0 + k0);          // K: k0 + hv*16 + 0..7
    b0.q[1] = *(const uint4*)(w0 + k0 + 8);      // K: k0 + hv*16 + 8..15
    b1.q[0] = *(const uint4*)(w1 + k0);
    b1.q[1] = *(const uint4*)(w1 + k0 + 8);
    b2.q[0] = *(const uint4*)(w2 + k0);
    b2.q[1] = *(const uint4*)(w2 + k0 + 8);
    b3.q[0] = *(const uint4*)(w3 + k0);
    b3.q[1] = *(const uint4*)(w3 + k0 + 8);

    acc0 = __builtin_amdgcn_wmma_f32_16x16x32_bf16(false, a.v, false, b0.v,
                                                   (short)0, acc0, false, false);
    acc1 = __builtin_amdgcn_wmma_f32_16x16x32_bf16(false, a.v, false, b1.v,
                                                   (short)0, acc1, false, false);
    acc2 = __builtin_amdgcn_wmma_f32_16x16x32_bf16(false, a.v, false, b2.v,
                                                   (short)0, acc2, false, false);
    acc3 = __builtin_amdgcn_wmma_f32_16x16x32_bf16(false, a.v, false, b3.v,
                                                   (short)0, acc3, false, false);
  }

  const float bi0 = bias ? bias[tn +  0 + lm] : 0.f;
  const float bi1 = bias ? bias[tn + 16 + lm] : 0.f;
  const float bi2 = bias ? bias[tn + 32 + lm] : 0.f;
  const float bi3 = bias ? bias[tn + 48 + lm] : 0.f;

#pragma unroll
  for (int r = 0; r < 8; ++r) {
    float* crow = C + (size_t)(tm + r + hv * 8) * N + tn;
    crow[ 0 + lm] = acc0[r] + bi0;
    crow[16 + lm] = acc1[r] + bi1;
    crow[32 + lm] = acc2[r] + bi2;
    crow[48 + lm] = acc3[r] + bi3;
  }
}

// ---------------------------------------------------------------------------
// Per bottom-pixel attention. One wave per (pixel, head); lane = head dim.
// For pixel (y,x): covering windows are (i,j) with r=y+1-2i in [0,4),
// c=x+1-2j in [0,4) -> <=4 windows, element e = r*4+c. All elements mapping
// to a pixel share one softmax denominator (the fold/unfold round trip).
// ---------------------------------------------------------------------------
__global__ void __launch_bounds__(256)
patch_attn(const float* __restrict__ Q,     // P x 256 (per-pixel projected q)
           const float* __restrict__ KV,    // T x 512 ([k(256) | v(256)])
           const float* __restrict__ rel_init,
           const float* __restrict__ rel_bias,
           unsigned short* __restrict__ O)  // P x 256 bf16
{
  const int id   = (int)((blockIdx.x * blockDim.x + threadIdx.x) >> 5);
  const int lane = (int)(threadIdx.x & 31u);
  const int h = id & 7;
  const int p = id >> 3;
  const int x  = p & 127;
  const int y  = (p >> 7) & 127;
  const int bb = p >> 14;

  const float qd = Q[(size_t)p * ATTN_DIM + h * ATTN_HD + lane];

  const int r0 = (y + 1) & 1;
  const int c0 = (x + 1) & 1;

  float ex[4];
  int   wid[4];
  float denom = 0.f;

#pragma unroll
  for (int a = 0; a < 2; ++a) {
    const int r = r0 + 2 * a;
    const int i = (y + 1 - r) >> 1;
    const bool vy = (i >= 0) && (i < 64);
#pragma unroll
    for (int b = 0; b < 2; ++b) {
      const int c = c0 + 2 * b;
      const int j = (x + 1 - c) >> 1;
      const bool ok = vy && (j >= 0) && (j < 64);
      const int idx = a * 2 + b;
      const int w = ok ? (bb * 4096 + i * 64 + j) : 0;
      wid[idx] = w;
      const float kd = ok ? KV[(size_t)w * ATTN_KVDIM + h * ATTN_HD + lane] : 0.f;
      float prod = qd * kd;
#pragma unroll
      for (int off = 16; off > 0; off >>= 1)
        prod += __shfl_xor(prod, off, 32);
      const int e = r * 4 + c;
      const float lg = prod * ATTN_SCALE + rel_init[e] + rel_bias[e];
      const float ev = ok ? __expf(lg) : 0.f;
      ex[idx] = ev;
      denom += ev;
    }
  }

  const float inv = 1.f / denom;   // always >= one positive exp
  float o = 0.f;
#pragma unroll
  for (int idx = 0; idx < 4; ++idx) {
    const float vv = KV[(size_t)wid[idx] * ATTN_KVDIM + 256 + h * ATTN_HD + lane];
    o += (ex[idx] * inv) * vv;     // ex==0 for invalid covers
  }
  O[(size_t)p * ATTN_DIM + h * ATTN_HD + lane] = f32_to_bf16_rne(o);
}

// ---------------------------------------------------------------------------
extern "C" void kernel_launch(void* const* d_in, const int* in_sizes, int n_in,
                              void* d_out, int out_size, void* d_ws, size_t ws_size,
                              hipStream_t stream) {
  const float* Xt       = (const float*)d_in[0];  // (B, 4096, 512)
  const float* Xb       = (const float*)d_in[1];  // (B, 16384, 256)
  const float* q_w      = (const float*)d_in[2];  // (256, 256)
  const float* kv_w     = (const float*)d_in[3];  // (512, 512)
  const float* proj_w   = (const float*)d_in[4];  // (256, 256)
  const float* proj_b   = (const float*)d_in[5];  // (256,)
  const float* rel_init = (const float*)d_in[6];  // (16,)
  const float* rel_bias = (const float*)d_in[7];  // (16,)

  const int B = in_sizes[1] / (128 * 128 * 256);
  const int P = B * 128 * 128;   // bottom pixels
  const int T = B * 64 * 64;     // coarse tokens

  // workspace carve-up (256B aligned)
  char* ws = (char*)d_ws;
  size_t off = 0;
  auto carve = [&](size_t bytes) -> char* {
    char* p = ws + off;
    off = (off + bytes + 255) & ~(size_t)255;
    return p;
  };
  unsigned short* xb_bf  = (unsigned short*)carve((size_t)P * 256 * 2);
  unsigned short* xt_bf  = (unsigned short*)carve((size_t)T * 512 * 2);
  unsigned short* qw_bf  = (unsigned short*)carve((size_t)256 * 256 * 2);
  unsigned short* kvw_bf = (unsigned short*)carve((size_t)512 * 512 * 2);
  unsigned short* pw_bf  = (unsigned short*)carve((size_t)256 * 256 * 2);
  float*          Qf     = (float*)carve((size_t)P * 256 * 4);
  float*          KVf    = (float*)carve((size_t)T * 512 * 4);
  unsigned short* o_bf   = (unsigned short*)carve((size_t)P * 256 * 2);
  (void)ws_size; (void)n_in; (void)out_size;

  // 1) fp32 -> bf16 for WMMA operands
  cvt_f32_to_bf16<<<4096, 256, 0, stream>>>(Xb, xb_bf, P * 256);
  cvt_f32_to_bf16<<<4096, 256, 0, stream>>>(Xt, xt_bf, T * 512);
  cvt_f32_to_bf16<<<256,  256, 0, stream>>>(q_w, qw_bf, 256 * 256);
  cvt_f32_to_bf16<<<256,  256, 0, stream>>>(kv_w, kvw_bf, 512 * 512);
  cvt_f32_to_bf16<<<256,  256, 0, stream>>>(proj_w, pw_bf, 256 * 256);

  // 2) Qpix = Xb @ q_w^T      (M=P, N=256, K=256)
  {
    int waves = (P / 16) * (256 / 64);
    gemm_bf16_wmma<<<waves / 8, 256, 0, stream>>>(xb_bf, qw_bf, nullptr, Qf,
                                                  P, 256, 256);
  }
  // 3) KV = Xt @ kv_w^T       (M=T, N=512, K=512)
  {
    int waves = (T / 16) * (512 / 64);
    gemm_bf16_wmma<<<waves / 8, 256, 0, stream>>>(xt_bf, kvw_bf, nullptr, KVf,
                                                  T, 512, 512);
  }
  // 4) per-pixel 4-way patch softmax attention -> o_bf
  patch_attn<<<P, 256, 0, stream>>>(Qf, KVf, rel_init, rel_bias, o_bf);

  // 5) out = o @ proj_w^T + proj_b  (M=P, N=256, K=256) -> d_out fp32
  {
    int waves = (P / 16) * (256 / 64);
    gemm_bf16_wmma<<<waves / 8, 256, 0, stream>>>(o_bf, pw_bf, proj_b,
                                                  (float*)d_out, P, 256, 256);
  }
}
